// GNL_70720931496668
// MI455X (gfx1250) — matrix operand order repaired and checked by reading, not compiled, not run on hardware
//
#include <hip/hip_runtime.h>

// ---------------------------------------------------------------------------
// Fused graph-attention GRU for MI455X (gfx1250).
// All GEMMs via v_wmma_f32_16x16x32_f16 (f32 accumulate).
// Round-3: adjacency packed to a bitmask (32x less L2 traffic on the dominant
// stream), softmax in exp2 domain (f1/f2 pre-scaled by log2 e), everything
// else keeps the b128 fragment data path from round 2.
// ---------------------------------------------------------------------------

typedef __attribute__((ext_vector_type(16))) _Float16 v16h;
typedef __attribute__((ext_vector_type(8)))  _Float16 v8h;
typedef __attribute__((ext_vector_type(8)))  float    v8f;

constexpr int kB = 8, kT = 12, kN = 2048, kH = 64, kCIN = 2;
constexpr int kNW = kN / 32;  // mask words per row
constexpr float kLog2e = 1.4426950408889634f;

#define DEV static __device__ __forceinline__

DEV v8f wmma_f16(v16h a, v16h b, v8f c) {
  // (neg_a, A, neg_b, B, c_mod, C, reuse_a, reuse_b)
  return __builtin_amdgcn_wmma_f32_16x16x32_f16(false, a, false, b, (short)0, c,
                                                false, false);
}

#if __has_builtin(__builtin_amdgcn_exp2f)
DEV float fast_exp2(float x) { return __builtin_amdgcn_exp2f(x); }
#else
DEV float fast_exp2(float x) { return __expf(x * 0.6931471805599453f); }
#endif

// Pre-swizzled B fragment: [tile][lane][16] contiguous f16 -> one v16h load.
DEV v16h load_frag(const _Float16* __restrict__ F, int tile, int lane) {
  return *(const v16h*)(F + (((size_t)tile << 5) + lane) * 16);
}

// A fragment (16x32 f16) from f32 row-major [rows x 64].
// ISA layout: lane half hi, element t -> K = kc*32 + hi*8 + (t<8 ? t : 8+t).
// Contiguous runs of 8 -> 4x float4 loads + pack-convert.
DEV v16h load_a_f32(const float* __restrict__ p, long row0, int m, int hi, int kc) {
  const float* rp = p + (row0 + m) * kH + kc * 32 + hi * 8;
  const float4 q0 = *(const float4*)(rp + 0);
  const float4 q1 = *(const float4*)(rp + 4);
  const float4 q2 = *(const float4*)(rp + 16);
  const float4 q3 = *(const float4*)(rp + 20);
  v16h a;
  a[0] = (_Float16)q0.x;  a[1] = (_Float16)q0.y;
  a[2] = (_Float16)q0.z;  a[3] = (_Float16)q0.w;
  a[4] = (_Float16)q1.x;  a[5] = (_Float16)q1.y;
  a[6] = (_Float16)q1.z;  a[7] = (_Float16)q1.w;
  a[8] = (_Float16)q2.x;  a[9] = (_Float16)q2.y;
  a[10] = (_Float16)q2.z; a[11] = (_Float16)q2.w;
  a[12] = (_Float16)q3.x; a[13] = (_Float16)q3.y;
  a[14] = (_Float16)q3.z; a[15] = (_Float16)q3.w;
  return a;
}

DEV float sigmoidf(float x) { return 1.f / (1.f + __expf(-x)); }
DEV float tanh_fast(float x) {
  x = fminf(fmaxf(x, -15.f), 15.f);
  float e = __expf(2.f * x);
  return (e - 1.f) / (e + 1.f);
}

// ---------------------------------------------------------------------------
// prep: H0 = X[:,0] @ W_h, weight fragment swizzle, adjacency bit-pack
// Fragment section = 2 kc-chunks x 4 col-tiles, layout [tile][lane][t]:
//   F[tile*512 + lane*16 + t] = W[(ro + kc*32 + hi*8 + (t<8 ? t : t+8)) * 64 + cc*16 + n]
// ---------------------------------------------------------------------------
DEV void fill_frag_section(const float* __restrict__ W, _Float16* __restrict__ F,
                           int ro, int tid, int stride) {
  for (int i = tid; i < 4096; i += stride) {
    const int tile = i >> 9;
    const int kc = tile >> 2, cc = tile & 3;
    const int lane = (i >> 4) & 31, tt = i & 15;
    const int hi = lane >> 4, n = lane & 15;
    const int k = ro + kc * 32 + hi * 8 + (tt < 8 ? tt : tt + 8);
    F[i] = (_Float16)W[(size_t)k * kH + cc * 16 + n];
  }
}

__global__ void k_prep(const float* __restrict__ X, const float* __restrict__ Wh,
                       const float* __restrict__ A, float* __restrict__ H,
                       unsigned* __restrict__ Abits,
                       const float* __restrict__ Wa, const float* __restrict__ Wf,
                       const float* __restrict__ We, const float* __restrict__ Wg,
                       const float* __restrict__ Wr, const float* __restrict__ Wu,
                       _Float16* __restrict__ FWa, _Float16* __restrict__ FWf,
                       _Float16* __restrict__ FWe, _Float16* __restrict__ FWg,
                       _Float16* __restrict__ FWr, _Float16* __restrict__ FWu) {
  const int tid = blockIdx.x * blockDim.x + threadIdx.x;
  const int stride = gridDim.x * blockDim.x;
  // H0 = X[:,0] @ W_h
  for (int row = tid; row < kB * kN; row += stride) {
    const int bb = row >> 11, nn = row & (kN - 1);
    const float* xp = X + (((size_t)bb * kT + 0) * kN + nn) * kCIN;
    const float x0 = xp[0], x1 = xp[1];
    for (int h = 0; h < kH; ++h)
      H[(size_t)row * kH + h] = x0 * Wh[h] + x1 * Wh[kH + h];
  }
  // adjacency bitmask: bit j of Abits[row*64 + w] = (A[row][w*32+j] > 0)
  for (int i = tid; i < kN * kNW; i += stride) {
    const int row = i >> 6, w = i & (kNW - 1);
    const float* ap = A + (size_t)row * kN + w * 32;
    unsigned bits = 0u;
#pragma unroll
    for (int q = 0; q < 32; q += 4) {
      const float4 v = *(const float4*)(ap + q);
      bits |= (v.x > 0.f ? 1u : 0u) << q;
      bits |= (v.y > 0.f ? 1u : 0u) << (q + 1);
      bits |= (v.z > 0.f ? 1u : 0u) << (q + 2);
      bits |= (v.w > 0.f ? 1u : 0u) << (q + 3);
    }
    Abits[i] = bits;
  }
  fill_frag_section(Wa, FWa, 0, tid, stride);
  fill_frag_section(Wf, FWf, 2, tid, stride);          // Z part (rows 2..65)
  fill_frag_section(Wf, FWf + 4096, 66, tid, stride);  // H part (rows 66..129)
  fill_frag_section(We, FWe, 2, tid, stride);
  fill_frag_section(We, FWe + 4096, 66, tid, stride);
  fill_frag_section(Wg, FWg, 2, tid, stride);
  fill_frag_section(Wg, FWg + 4096, 66, tid, stride);
  fill_frag_section(Wr, FWr, 2, tid, stride);
  fill_frag_section(Wr, FWr + 4096, 66, tid, stride);
  fill_frag_section(Wu, FWu, 2, tid, stride);
  fill_frag_section(Wu, FWu + 4096, 66, tid, stride);
}

// ---------------------------------------------------------------------------
// Hz = H @ W_alpha (WMMA) -> HzT16 (transposed, [b][col][j], v8h stores).
// Fused: f1 = Hz@a_src, f2 = Hz@a_dst straight from the accumulators via
// 16-lane shfl_xor reductions. f1/f2 stored pre-scaled by log2(e) so the
// attention kernel's softmax is a bare v_exp_f32 (exp2).
// ---------------------------------------------------------------------------
__global__ __launch_bounds__(128) void k_hz(const float* __restrict__ H,
                                            const _Float16* __restrict__ FWa,
                                            const float* __restrict__ av,
                                            _Float16* __restrict__ HzT,
                                            float* __restrict__ f1,
                                            float* __restrict__ f2,
                                            unsigned* __restrict__ f2key) {
  if (blockIdx.x == 0 && threadIdx.x < kB) f2key[threadIdx.x] = 0u;
  const int wave = threadIdx.x >> 5, lane = threadIdx.x & 31;
  const int hi = lane >> 4, m = lane & 15;
  const long row0 = ((long)blockIdx.x * 4 + wave) * 16;
  const int b = (int)(row0 >> 11);
  const int n0 = (int)(row0 & (kN - 1));
  v8f acc[4] = {};
#pragma unroll
  for (int kc = 0; kc < 2; ++kc) {
    const v16h a = load_a_f32(H, row0, m, hi, kc);
#pragma unroll
    for (int cc = 0; cc < 4; ++cc)
      acc[cc] = wmma_f16(a, load_frag(FWa, kc * 4 + cc, lane), acc[cc]);
  }
  // transposed f16 store: HzT[(b*64 + col) * kN + j], 8 consecutive j per pack
#pragma unroll
  for (int cc = 0; cc < 4; ++cc) {
    v8h pk;
#pragma unroll
    for (int r = 0; r < 8; ++r) pk[r] = (_Float16)acc[cc][r];
    *(v8h*)(HzT + (size_t)(b * kH + cc * 16 + m) * kN + n0 + hi * 8) = pk;
  }
  // fused f1/f2 (log2e-scaled): per-lane partial, reduce across 16 lanes
  const float as0 = av[m],      as1 = av[16 + m],      as2 = av[32 + m],      as3 = av[48 + m];
  const float ad0 = av[64 + m], ad1 = av[80 + m],      ad2 = av[96 + m],      ad3 = av[112 + m];
  float mx = -3.0e38f;
#pragma unroll
  for (int r = 0; r < 8; ++r) {
    float s1 = acc[0][r] * as0 + acc[1][r] * as1 + acc[2][r] * as2 + acc[3][r] * as3;
    float s2 = acc[0][r] * ad0 + acc[1][r] * ad1 + acc[2][r] * ad2 + acc[3][r] * ad3;
#pragma unroll
    for (int msk = 1; msk < 16; msk <<= 1) {
      s1 += __shfl_xor(s1, msk, 32);
      s2 += __shfl_xor(s2, msk, 32);
    }
    if (m == 0) {
      const long row = row0 + r + hi * 8;
      s1 *= kLog2e;
      s2 *= kLog2e;
      f1[row] = s1;
      f2[row] = s2;
      mx = fmaxf(mx, s2);
    }
  }
  if (m == 0) {
    const unsigned bits = __float_as_uint(mx);
    const unsigned key = (bits & 0x80000000u) ? ~bits : (bits | 0x80000000u);
    atomicMax(&f2key[b], key);
  }
}

// ---------------------------------------------------------------------------
// Fused masked softmax + Z = att @ Hz.  Grid (N/64, B), 4 waves x 16 rows.
// Scores in exp2 domain; leakyrelu commutes with the positive log2e scale, so
// M_i = leakyrelu(f1'_i + max_b f2') still upper-bounds the row max.
// Mask read as 1 u32/lane/chunk; B fragments as two 16B v8h loads from HzT16.
// ---------------------------------------------------------------------------
__global__ __launch_bounds__(128) void k_att(const unsigned* __restrict__ Abits,
                                             const float* __restrict__ f1,
                                             const float* __restrict__ f2,
                                             const unsigned* __restrict__ f2key,
                                             const _Float16* __restrict__ HzT,
                                             float* __restrict__ Z) {
  __shared__ float sRS[4][16];
  const int b = blockIdx.y;
  const int wave = threadIdx.x >> 5, lane = threadIdx.x & 31;
  const int hi = lane >> 4, m = lane & 15;
  const long base = (long)b * kN;
  const int i0 = blockIdx.x * 64 + wave * 16;
  const float f1m = f1[base + i0 + m];
  const unsigned key = f2key[b];
  const float f2max = (key & 0x80000000u) ? __uint_as_float(key ^ 0x80000000u)
                                          : __uint_as_float(~key);
  float M = f1m + f2max;
  M = (M > 0.f) ? M : 0.2f * M;
  const unsigned* abrow = Abits + (size_t)(i0 + m) * kNW;  // this lane's mask row
  const float* f2p = f2 + base + hi * 8;
  const _Float16* hzb = HzT + (size_t)b * kH * kN;
  v8f acc[4] = {};
  float rsum = 0.f;

  for (int j0 = 0; j0 < kN; j0 += 32) {
    const unsigned aw = abrow[j0 >> 5];
    const float4 g0 = *(const float4*)(f2p + j0);
    const float4 g1 = *(const float4*)(f2p + j0 + 4);
    const float4 g2 = *(const float4*)(f2p + j0 + 16);
    const float4 g3 = *(const float4*)(f2p + j0 + 20);
    float fm[16] = {g0.x, g0.y, g0.z, g0.w, g1.x, g1.y, g1.z, g1.w,
                    g2.x, g2.y, g2.z, g2.w, g3.x, g3.y, g3.z, g3.w};
    v16h pa;
#pragma unroll
    for (int t = 0; t < 16; ++t) {
      const int kb = (t < 8) ? (hi * 8 + t) : (8 + hi * 8 + t);
      float s = f1m + fm[t];
      s = (s > 0.f) ? s : 0.2f * s;
      const float p = ((aw >> kb) & 1u) ? fast_exp2(s - M) : 0.f;
      rsum += p;
      pa[t] = (_Float16)p;
    }
#pragma unroll
    for (int cc = 0; cc < 4; ++cc) {
      const _Float16* bp = hzb + (size_t)(cc * 16 + m) * kN + j0 + hi * 8;
      const v8h b0 = *(const v8h*)bp;
      const v8h b1 = *(const v8h*)(bp + 16);
      const v16h bf = __builtin_shufflevector(b0, b1, 0, 1, 2, 3, 4, 5, 6, 7, 8, 9,
                                              10, 11, 12, 13, 14, 15);
      acc[cc] = wmma_f16(pa, bf, acc[cc]);
    }
  }
  rsum += __shfl_xor(rsum, 16, 32);  // combine the two K-halves of each row
  if (hi == 0) sRS[wave][m] = rsum;
  __syncthreads();
#pragma unroll
  for (int cc = 0; cc < 4; ++cc)
#pragma unroll
    for (int r = 0; r < 8; ++r) {
      const long row = base + i0 + r + hi * 8;
      const float inv = 1.f / sRS[wave][r + hi * 8];
      Z[row * kH + cc * 16 + m] = acc[cc][r] * inv;
    }
}

// ---------------------------------------------------------------------------
// Gates + state update. cat@W split: x-part (K=2) in VALU, Z/H parts (K=64
// each) via WMMA with pre-swizzled fragments. u() factored:
// u = tanh(ax + zz@Wu_z + hh@Wu_h); ax folded into the hh partials.
// Z_hat/H_hat round-trip through per-wave LDS to rebuild A-layout fragments.
// ---------------------------------------------------------------------------
__global__ __launch_bounds__(128) void k_gates(
    const float* __restrict__ X, int t, const float* __restrict__ Z,
    float* __restrict__ H, const _Float16* __restrict__ FWf,
    const _Float16* __restrict__ FWe, const _Float16* __restrict__ FWg,
    const _Float16* __restrict__ FWr, const _Float16* __restrict__ FWu,
    const float* __restrict__ Wf, const float* __restrict__ We,
    const float* __restrict__ Wg, const float* __restrict__ Wr,
    const float* __restrict__ Wu) {
  __shared__ _Float16 sZh[4][16][kH];
  __shared__ _Float16 sHh[4][16][kH];
  const int wave = threadIdx.x >> 5, lane = threadIdx.x & 31;
  const int hi = lane >> 4, m = lane & 15;
  const long row0 = ((long)blockIdx.x * 4 + wave) * 16;

  v8f Sf[4] = {}, Se[4] = {}, Sg[4] = {}, Sr[4] = {};
  v8f Pz[4] = {}, Ph[4] = {}, Phh[4] = {};
#pragma unroll
  for (int kc = 0; kc < 2; ++kc) {
    const v16h aZ = load_a_f32(Z, row0, m, hi, kc);
    const v16h aH = load_a_f32(H, row0, m, hi, kc);
#pragma unroll
    for (int cc = 0; cc < 4; ++cc) {
      const int tz = kc * 4 + cc, th = 8 + kc * 4 + cc;
      Sf[cc] = wmma_f16(aZ, load_frag(FWf, tz, lane), Sf[cc]);
      Sf[cc] = wmma_f16(aH, load_frag(FWf, th, lane), Sf[cc]);
      Se[cc] = wmma_f16(aZ, load_frag(FWe, tz, lane), Se[cc]);
      Se[cc] = wmma_f16(aH, load_frag(FWe, th, lane), Se[cc]);
      Sg[cc] = wmma_f16(aZ, load_frag(FWg, tz, lane), Sg[cc]);
      Sg[cc] = wmma_f16(aH, load_frag(FWg, th, lane), Sg[cc]);
      Sr[cc] = wmma_f16(aZ, load_frag(FWr, tz, lane), Sr[cc]);
      Sr[cc] = wmma_f16(aH, load_frag(FWr, th, lane), Sr[cc]);
      Pz[cc] = wmma_f16(aZ, load_frag(FWu, tz, lane), Pz[cc]);
      Ph[cc] = wmma_f16(aH, load_frag(FWu, th, lane), Ph[cc]);
    }
  }
  // elementwise phase 1: add x-part, compute Z_hat/H_hat -> LDS, g/r sigmoids
#pragma unroll
  for (int r = 0; r < 8; ++r) {
    const long row = row0 + r + hi * 8;
    const int bb = (int)(row >> 11), nn = (int)(row & (kN - 1));
    const float* xp = X + (((size_t)bb * kT + t) * kN + nn) * kCIN;
    const float x0 = xp[0], x1 = xp[1];
#pragma unroll
    for (int cc = 0; cc < 4; ++cc) {
      const int col = cc * 16 + m;
      const float sf = Sf[cc][r] + x0 * Wf[col] + x1 * Wf[kH + col];
      const float se = Se[cc][r] + x0 * We[col] + x1 * We[kH + col];
      const float sg = Sg[cc][r] + x0 * Wg[col] + x1 * Wg[kH + col];
      const float sr = Sr[cc][r] + x0 * Wr[col] + x1 * Wr[kH + col];
      const float ax = x0 * Wu[col] + x1 * Wu[kH + col];
      Ph[cc][r] += ax;   // ax folded into hid partial
      Phh[cc][r] = ax;   // and seeds the H_hat partial
      const float zv = Z[row * kH + col];
      const float hv = H[row * kH + col];
      sZh[wave][r + hi * 8][col] = (_Float16)(sigmoidf(sf) * zv);
      sHh[wave][r + hi * 8][col] = (_Float16)(sigmoidf(se) * hv);
      Sg[cc][r] = sigmoidf(sg);
      Sr[cc][r] = sigmoidf(sr);
    }
  }
  __syncthreads();
  // phase 2: Pzh = Z_hat@Wu_z, Phh += H_hat@Wu_h (A frags from LDS, runs of 8)
  v8f Pzh[4] = {};
#pragma unroll
  for (int kc = 0; kc < 2; ++kc) {
    const _Float16* zp = &sZh[wave][m][kc * 32 + hi * 8];
    const _Float16* hp = &sHh[wave][m][kc * 32 + hi * 8];
    v16h aZh, aHh;
#pragma unroll
    for (int tt = 0; tt < 8; ++tt) {
      aZh[tt] = zp[tt];
      aZh[8 + tt] = zp[16 + tt];
      aHh[tt] = hp[tt];
      aHh[8 + tt] = hp[16 + tt];
    }
#pragma unroll
    for (int cc = 0; cc < 4; ++cc) {
      Pzh[cc] = wmma_f16(aZh, load_frag(FWu, kc * 4 + cc, lane), Pzh[cc]);
      Phh[cc] = wmma_f16(aHh, load_frag(FWu, 8 + kc * 4 + cc, lane), Phh[cc]);
    }
  }
  // final combine: H_new (in-place; rows are wave-exclusive)
#pragma unroll
  for (int cc = 0; cc < 4; ++cc)
#pragma unroll
    for (int r = 0; r < 8; ++r) {
      const long row = row0 + r + hi * 8;
      const int col = cc * 16 + m;
      const float u1 = tanh_fast(Pzh[cc][r] + Phh[cc][r]);
      const float u2 = tanh_fast(Pz[cc][r] + Phh[cc][r]);
      const float u3 = tanh_fast(Pzh[cc][r] + Ph[cc][r]);
      const float u4 = tanh_fast(Pz[cc][r] + Ph[cc][r]);
      const float g = Sg[cc][r], rr = Sr[cc][r];
      H[row * kH + col] = g * rr * u1 + (1.f - g) * rr * u2 +
                          g * (1.f - rr) * u3 + (1.f - g) * (1.f - rr) * u4;
    }
}

// ---------------------------------------------------------------------------
// out = Hf @ fc_w.T + fc_b
// ---------------------------------------------------------------------------
__global__ void k_out(const float* __restrict__ H, const float* __restrict__ fcw,
                      const float* __restrict__ fcb, float* __restrict__ out) {
  const int row = blockIdx.x * blockDim.x + threadIdx.x;
  if (row >= kB * kN) return;
  float s0 = fcb[0], s1 = fcb[1];
#pragma unroll
  for (int h4 = 0; h4 < kH; h4 += 4) {
    const float4 hv = *(const float4*)(H + (size_t)row * kH + h4);
    s0 += hv.x * fcw[h4] + hv.y * fcw[h4 + 1] + hv.z * fcw[h4 + 2] + hv.w * fcw[h4 + 3];
    s1 += hv.x * fcw[kH + h4] + hv.y * fcw[kH + h4 + 1] + hv.z * fcw[kH + h4 + 2] +
          hv.w * fcw[kH + h4 + 3];
  }
  out[row * 2 + 0] = s0;
  out[row * 2 + 1] = s1;
}

// ---------------------------------------------------------------------------
// Workspace layout (bytes)
// ---------------------------------------------------------------------------
constexpr size_t OFF_H    = 0;
constexpr size_t OFF_Z    = OFF_H + (size_t)kB * kN * kH * 4;
constexpr size_t OFF_HZT  = OFF_Z + (size_t)kB * kN * kH * 4;
constexpr size_t OFF_F1   = OFF_HZT + (size_t)kB * kN * kH * 2;
constexpr size_t OFF_F2   = OFF_F1 + (size_t)kB * kN * 4;
constexpr size_t OFF_KEY  = OFF_F2 + (size_t)kB * kN * 4;
constexpr size_t OFF_FWA  = OFF_KEY + 256;
constexpr size_t OFF_FWF  = OFF_FWA + (size_t)4096 * 2;
constexpr size_t OFF_FWE  = OFF_FWF + (size_t)8192 * 2;
constexpr size_t OFF_FWG  = OFF_FWE + (size_t)8192 * 2;
constexpr size_t OFF_FWR  = OFF_FWG + (size_t)8192 * 2;
constexpr size_t OFF_FWU  = OFF_FWR + (size_t)8192 * 2;
constexpr size_t OFF_AB   = OFF_FWU + (size_t)8192 * 2;  // 2048*64 u32 = 512KB

extern "C" void kernel_launch(void* const* d_in, const int* in_sizes, int n_in,
                              void* d_out, int out_size, void* d_ws, size_t ws_size,
                              hipStream_t stream) {
  const float* X   = (const float*)d_in[0];
  const float* A   = (const float*)d_in[1];
  const float* Wh  = (const float*)d_in[2];
  const float* Wa  = (const float*)d_in[3];
  const float* av  = (const float*)d_in[4];
  const float* Wf  = (const float*)d_in[5];
  const float* We  = (const float*)d_in[6];
  const float* Wg  = (const float*)d_in[7];
  const float* Wr  = (const float*)d_in[8];
  const float* Wu  = (const float*)d_in[9];
  const float* fcw = (const float*)d_in[10];
  const float* fcb = (const float*)d_in[11];

  char* ws = (char*)d_ws;
  float*     Hc    = (float*)(ws + OFF_H);
  float*     Zbuf  = (float*)(ws + OFF_Z);
  _Float16*  HzT   = (_Float16*)(ws + OFF_HZT);
  float*     f1    = (float*)(ws + OFF_F1);
  float*     f2    = (float*)(ws + OFF_F2);
  unsigned*  f2key = (unsigned*)(ws + OFF_KEY);
  _Float16*  FWa   = (_Float16*)(ws + OFF_FWA);
  _Float16*  FWf   = (_Float16*)(ws + OFF_FWF);
  _Float16*  FWe   = (_Float16*)(ws + OFF_FWE);
  _Float16*  FWg   = (_Float16*)(ws + OFF_FWG);
  _Float16*  FWr   = (_Float16*)(ws + OFF_FWR);
  _Float16*  FWu   = (_Float16*)(ws + OFF_FWU);
  unsigned*  Abits = (unsigned*)(ws + OFF_AB);

  k_prep<<<256, 256, 0, stream>>>(X, Wh, A, Hc, Abits, Wa, Wf, We, Wg, Wr, Wu,
                                  FWa, FWf, FWe, FWg, FWr, FWu);

  for (int t = 0; t < kT; ++t) {
    k_hz<<<256, 128, 0, stream>>>(Hc, FWa, av, HzT, f1, f2, f2key);
    k_att<<<dim3(kN / 64, kB), 128, 0, stream>>>(Abits, f1, f2, f2key, HzT, Zbuf);
    k_gates<<<256, 128, 0, stream>>>(X, t, Zbuf, Hc, FWf, FWe, FWg, FWr, FWu,
                                     Wf, We, Wg, Wr, Wu);
  }
  k_out<<<64, 256, 0, stream>>>(Hc, fcw, fcb, (float*)d_out);
}
